// SAGS_GCN_70944269796063
// MI455X (gfx1250) — matrix operand order repaired and checked by reading, not compiled, not run on hardware
//
#include <hip/hip_runtime.h>
#include <cstdint>
#include <cstddef>

// ---------------------------------------------------------------------------
// SAGS-GCN for MI455X (gfx1250, wave32).
// Heavy work = two 4096^3 GEMMs + two 4096x4096x512 GEMMs, all through
// v_wmma_f32_16x16x32_bf16 (f32 accumulate). B tile is stored transposed in
// LDS (with an XOR bank swizzle) so both A and B fragments load as
// ds_load_b128; LDS tiles are double-buffered with register staging so each
// K-iteration has a single barrier and global loads overlap the WMMA work.
// The staged tile is held in a plain struct of uint4 scalars (no local
// arrays, no lambdas) so it can never be demoted to a scratch stack slot.
// ---------------------------------------------------------------------------

typedef __bf16 bf16;
typedef __bf16 v16bf __attribute__((ext_vector_type(16)));
typedef __bf16 v8bf  __attribute__((ext_vector_type(8)));
typedef float  v8f   __attribute__((ext_vector_type(8)));

static __device__ __forceinline__ unsigned short bfbits(float f) {
    return __builtin_bit_cast(unsigned short, (bf16)f);
}
static __device__ __forceinline__ v16bf cat8(v8bf lo, v8bf hi) {
    return __builtin_shufflevector(lo, hi, 0, 1, 2, 3, 4, 5, 6, 7,
                                   8, 9, 10, 11, 12, 13, 14, 15);
}

// ---------------------------------------------------------------------------
// f32 -> bf16 convert (graph matrix)
// ---------------------------------------------------------------------------
__global__ void k_f32_to_bf16(const float* __restrict__ in,
                              unsigned short* __restrict__ out, int n) {
    int stride = gridDim.x * blockDim.x;
    for (int i = blockIdx.x * blockDim.x + threadIdx.x; i < n; i += stride)
        out[i] = bfbits(in[i]);
}

// ---------------------------------------------------------------------------
// Build XT[m, b*C + c] = x[b, m, c] as bf16  (x is [B,N,C] f32)
// ---------------------------------------------------------------------------
__global__ void k_build_xt(const float* __restrict__ x,
                           unsigned short* __restrict__ xt,
                           int B, int N, int C) {
    int total = B * N * C;
    int stride = gridDim.x * blockDim.x;
    for (int i = blockIdx.x * blockDim.x + threadIdx.x; i < total; i += stride) {
        int c = i % C;
        int b = (i / C) % B;
        int m = i / (C * B);
        xt[i] = bfbits(x[((size_t)b * N + m) * C + c]);
    }
}

// ---------------------------------------------------------------------------
// P[row, :] = softmax(relu(E[row] . E[:]^T))  -> bf16
// One block (256 threads) per row; D = 16.
// ---------------------------------------------------------------------------
__global__ __launch_bounds__(256)
void k_adjacency(const float* __restrict__ E,
                 unsigned short* __restrict__ P, int N) {
    const int row = blockIdx.x;
    const int tid = threadIdx.x;
    __shared__ float se[16];
    __shared__ float red[256];

    if (tid < 16) se[tid] = E[(size_t)row * 16 + tid];
    __syncthreads();

    const int PER = N / 256;   // 16
    float loc[16];
    float mx = 0.f;            // relu output >= 0, so 0 is a valid floor
    for (int j = 0; j < PER; ++j) {
        int m = tid + j * 256;
        const float4* ep = reinterpret_cast<const float4*>(E + (size_t)m * 16);
        float d = 0.f;
#pragma unroll
        for (int q = 0; q < 4; ++q) {
            float4 v = ep[q];
            d += v.x * se[q * 4 + 0] + v.y * se[q * 4 + 1] +
                 v.z * se[q * 4 + 2] + v.w * se[q * 4 + 3];
        }
        d = fmaxf(d, 0.f);
        loc[j] = d;
        mx = fmaxf(mx, d);
    }
    red[tid] = mx;
    __syncthreads();
    for (int s = 128; s > 0; s >>= 1) {
        if (tid < s) red[tid] = fmaxf(red[tid], red[tid + s]);
        __syncthreads();
    }
    mx = red[0];
    __syncthreads();

    float sum = 0.f;
    for (int j = 0; j < PER; ++j) {
        float v = __expf(loc[j] - mx);
        loc[j] = v;
        sum += v;
    }
    red[tid] = sum;
    __syncthreads();
    for (int s = 128; s > 0; s >>= 1) {
        if (tid < s) red[tid] += red[tid + s];
        __syncthreads();
    }
    const float inv = 1.f / red[0];
    for (int j = 0; j < PER; ++j)
        P[(size_t)row * N + tid + j * 256] = bfbits(loc[j] * inv);
}

// ---------------------------------------------------------------------------
// Tiled bf16 GEMM with f32 accumulate:
//   C = alpha * (A @ B)  [- I if sub_identity]
// A: [M,K] bf16 row-major, B: [K,N] bf16 row-major.
// Block tile 128x128x64, 8 waves (2 M x 4 N), wave tile 64x32 = 4x2 WMMA
// tiles -> v_wmma_f32_16x16x32_bf16.
//
// LDS:
//   lds_a[buf][m][k]  row-major, 16B-aligned rows  (A frag = 2x ds_load_b128)
//   lds_b[buf][n][k]  TRANSPOSED, with the 8-elem K-group XOR-swizzled by
//                     (n>>3)&7 to spread transposed b32 stores over banks.
//                     (B frag = 2x ds_load_b128)
// Double buffered, register-staged: one barrier per K-iteration.
// ---------------------------------------------------------------------------
#define BM 128
#define BN 128
#define BK 64
#define LDSA 72    // BK + 8 pad  (row = 144 B, multiple of 16)
#define LDTB 72    // BK + 8 pad for transposed B rows

// Staged next tile: plain scalars only (promotable to VGPRs, never stack).
struct StageRegs {
    uint4 a0, a1, a2, a3;     // A tile: 4 x (8 bf16 along K)
    uint4 be0, bo0;           // B tile slot 0: K-even / K-odd rows
    uint4 be1, bo1;           // B tile slot 1
};

static __device__ __forceinline__ uint4 ldg4(const unsigned short* p) {
    return *reinterpret_cast<const uint4*>(p);
}

static __device__ __forceinline__ StageRegs
stage_tile(const unsigned short* __restrict__ A,
           const unsigned short* __restrict__ Bm,
           int tid, int bm0, int bn0, int K, int N, int k0) {
    StageRegs s;
    // A: u = tid + j*256; r = u>>3; c8 = u&7
    s.a0 = ldg4(A + (size_t)(bm0 + ((tid +   0) >> 3)) * K + k0 + ((tid +   0) & 7) * 8);
    s.a1 = ldg4(A + (size_t)(bm0 + ((tid + 256) >> 3)) * K + k0 + ((tid + 256) & 7) * 8);
    s.a2 = ldg4(A + (size_t)(bm0 + ((tid + 512) >> 3)) * K + k0 + ((tid + 512) & 7) * 8);
    s.a3 = ldg4(A + (size_t)(bm0 + ((tid + 768) >> 3)) * K + k0 + ((tid + 768) & 7) * 8);
    // B: u = tid + j*256; r2 = u>>4 (K pair); c8 = u&15 (N group)
    {
        int r2 = tid >> 4, c8 = tid & 15;
        s.be0 = ldg4(Bm + (size_t)(k0 + 2 * r2)     * N + bn0 + c8 * 8);
        s.bo0 = ldg4(Bm + (size_t)(k0 + 2 * r2 + 1) * N + bn0 + c8 * 8);
    }
    {
        int u = tid + 256;
        int r2 = u >> 4, c8 = u & 15;
        s.be1 = ldg4(Bm + (size_t)(k0 + 2 * r2)     * N + bn0 + c8 * 8);
        s.bo1 = ldg4(Bm + (size_t)(k0 + 2 * r2 + 1) * N + bn0 + c8 * 8);
    }
    return s;
}

// Interleave one dword of the K-even row with one dword of the K-odd row and
// store the two resulting (k,k+1) bf16 pairs transposed into LDS.
static __device__ __forceinline__ void
put2(unsigned short* __restrict__ ldsb, int r2, int n_even,
     unsigned int ev, unsigned int od) {
    unsigned int lo = (ev & 0xFFFFu) | (od << 16);          // column n_even
    unsigned int hi = (ev >> 16) | (od & 0xFFFF0000u);      // column n_even+1
    const int n_odd = n_even + 1;
    const int pge = ((r2 >> 2) ^ ((n_even >> 3) & 7));
    const int pgo = ((r2 >> 2) ^ ((n_odd  >> 3) & 7));
    *reinterpret_cast<unsigned int*>(
        &ldsb[n_even * LDTB + pge * 8 + 2 * (r2 & 3)]) = lo;
    *reinterpret_cast<unsigned int*>(
        &ldsb[n_odd  * LDTB + pgo * 8 + 2 * (r2 & 3)]) = hi;
}

static __device__ __forceinline__ void
commit_tile(const StageRegs& s, unsigned short* __restrict__ ldsa,
            unsigned short* __restrict__ ldsb, int tid) {
    *reinterpret_cast<uint4*>(&ldsa[((tid +   0) >> 3) * LDSA + ((tid +   0) & 7) * 8]) = s.a0;
    *reinterpret_cast<uint4*>(&ldsa[((tid + 256) >> 3) * LDSA + ((tid + 256) & 7) * 8]) = s.a1;
    *reinterpret_cast<uint4*>(&ldsa[((tid + 512) >> 3) * LDSA + ((tid + 512) & 7) * 8]) = s.a2;
    *reinterpret_cast<uint4*>(&ldsa[((tid + 768) >> 3) * LDSA + ((tid + 768) & 7) * 8]) = s.a3;
    {
        int r2 = tid >> 4, n0 = (tid & 15) * 8;
        put2(ldsb, r2, n0 + 0, s.be0.x, s.bo0.x);
        put2(ldsb, r2, n0 + 2, s.be0.y, s.bo0.y);
        put2(ldsb, r2, n0 + 4, s.be0.z, s.bo0.z);
        put2(ldsb, r2, n0 + 6, s.be0.w, s.bo0.w);
    }
    {
        int u = tid + 256;
        int r2 = u >> 4, n0 = (u & 15) * 8;
        put2(ldsb, r2, n0 + 0, s.be1.x, s.bo1.x);
        put2(ldsb, r2, n0 + 2, s.be1.y, s.bo1.y);
        put2(ldsb, r2, n0 + 4, s.be1.z, s.bo1.z);
        put2(ldsb, r2, n0 + 6, s.be1.w, s.bo1.w);
    }
}

template <bool OUT_BF16>
__global__ __launch_bounds__(256)
__attribute__((amdgpu_waves_per_eu(2)))
void k_gemm_bf16(const unsigned short* __restrict__ A,
                 const unsigned short* __restrict__ Bm,
                 void* __restrict__ Cout,
                 int M, int N, int K, float alpha, int sub_identity) {
    __shared__ unsigned short lds_a[2][BM * LDSA];
    __shared__ unsigned short lds_b[2][BN * LDTB];

    const int tid   = threadIdx.x;
    const int lane  = tid & 31;
    const int wave  = tid >> 5;     // 0..7
    const int waveM = wave & 1;     // 64 rows per wave
    const int waveN = wave >> 1;    // 32 cols per wave
    const int hl    = lane >> 4;    // half-wave select
    const int l16   = lane & 15;

    const int bm0 = blockIdx.y * BM;
    const int bn0 = blockIdx.x * BN;

    v8f acc[4][2];
#pragma unroll
    for (int mt = 0; mt < 4; ++mt)
#pragma unroll
        for (int nt = 0; nt < 2; ++nt)
#pragma unroll
            for (int r = 0; r < 8; ++r) acc[mt][nt][r] = 0.f;

    StageRegs sreg = stage_tile(A, Bm, tid, bm0, bn0, K, N, 0);
    int p = 0;
    for (int k0 = 0; k0 < K; k0 += BK, p ^= 1) {
        commit_tile(sreg, &lds_a[p][0], &lds_b[p][0], tid);
        __syncthreads();
        if (k0 + BK < K)   // overlap next global loads with the WMMA work
            sreg = stage_tile(A, Bm, tid, bm0, bn0, K, N, k0 + BK);

#pragma unroll
        for (int ks = 0; ks < BK; ks += 32) {
            // ---- B fragments first (stay live), then stream A per mt.
            const int kbB = ks + hl * 16;
            v16bf b0, b1;
            {
                const int n  = waveN * 32 + l16;
                const int sw = (n >> 3) & 7;
                v8bf blo = *reinterpret_cast<const v8bf*>(
                    &lds_b[p][n * LDTB + (((kbB)     >> 3) ^ sw) * 8]);
                v8bf bhi = *reinterpret_cast<const v8bf*>(
                    &lds_b[p][n * LDTB + (((kbB + 8) >> 3) ^ sw) * 8]);
                b0 = cat8(blo, bhi);
            }
            {
                const int n  = waveN * 32 + 16 + l16;
                const int sw = (n >> 3) & 7;
                v8bf blo = *reinterpret_cast<const v8bf*>(
                    &lds_b[p][n * LDTB + (((kbB)     >> 3) ^ sw) * 8]);
                v8bf bhi = *reinterpret_cast<const v8bf*>(
                    &lds_b[p][n * LDTB + (((kbB + 8) >> 3) ^ sw) * 8]);
                b1 = cat8(blo, bhi);
            }
#pragma unroll
            for (int mt = 0; mt < 4; ++mt) {
                const int m  = waveM * 64 + mt * 16 + l16;
                const int kb = ks + hl * 16;
                v8bf alo = *reinterpret_cast<const v8bf*>(&lds_a[p][m * LDSA + kb]);
                v8bf ahi = *reinterpret_cast<const v8bf*>(&lds_a[p][m * LDSA + kb + 8]);
                v16bf a = cat8(alo, ahi);
                acc[mt][0] = __builtin_amdgcn_wmma_f32_16x16x32_bf16(
                    false, a, false, b0, (short)0, acc[mt][0], false, false);
                acc[mt][1] = __builtin_amdgcn_wmma_f32_16x16x32_bf16(
                    false, a, false, b1, (short)0, acc[mt][1], false, false);
            }
        }
        // Single barrier per iteration: next commit() writes the OTHER buffer;
        // buffer p is rewritten only at k0+2*BK, i.e. after the next barrier.
    }

    // ---- store: C/D layout — lanes 0-15 M=r, lanes 16-31 M=8+r; N=lane&15
#pragma unroll
    for (int mt = 0; mt < 4; ++mt) {
#pragma unroll
        for (int nt = 0; nt < 2; ++nt) {
            const int n = bn0 + waveN * 32 + nt * 16 + l16;
#pragma unroll
            for (int r = 0; r < 8; ++r) {
                const int m = bm0 + waveM * 64 + mt * 16 + hl * 8 + r;
                float v = alpha * acc[mt][nt][r];
                if (sub_identity && m == n) v -= 1.f;
                if (OUT_BF16)
                    reinterpret_cast<unsigned short*>(Cout)[(size_t)m * N + n] = bfbits(v);
                else
                    reinterpret_cast<float*>(Cout)[(size_t)m * N + n] = v;
            }
        }
    }
}

// ---------------------------------------------------------------------------
// Final per-node contraction:
//   W[n,k,i,o] = sum_d E[n,d] * Wp[d,k,i,o]   (in LDS)
//   out[b,n,o] = bias[n,o] + sum_i x[b,n,i]*W[n,0,i,o]
//                           + Xg1[n,b*C+i]*W[n,1,i,o]
//                           + Xg2[n,b*C+i]*W[n,2,i,o]
// One block (512 threads = 16 B x 32 Cout) per node n.
// ---------------------------------------------------------------------------
__global__ __launch_bounds__(512)
void k_final(const float* __restrict__ x,
             const float* __restrict__ E,
             const float* __restrict__ Wp,
             const float* __restrict__ Bp,
             const float* __restrict__ Xg1,
             const float* __restrict__ Xg2,
             float* __restrict__ out,
             int Bn, int N, int C) {
    const int n   = blockIdx.x;
    const int tid = threadIdx.x;
    __shared__ float se[16];
    __shared__ float W[3 * 32 * 32];
    __shared__ float bias[32];

    if (tid < 16) se[tid] = E[(size_t)n * 16 + tid];
    __syncthreads();

    for (int idx = tid; idx < 3072; idx += 512) {
        float a = 0.f;
#pragma unroll
        for (int d = 0; d < 16; ++d) a += se[d] * Wp[(size_t)d * 3072 + idx];
        W[idx] = a;
    }
    if (tid < 32) {
        float a = 0.f;
#pragma unroll
        for (int d = 0; d < 16; ++d) a += se[d] * Bp[d * 32 + tid];
        bias[tid] = a;
    }
    __syncthreads();

    const int b = tid >> 5;
    const int o = tid & 31;
    const float* x0 = x   + ((size_t)b * N + n) * C;
    const float* g1 = Xg1 + (size_t)n * (size_t)(Bn * C) + b * C;
    const float* g2 = Xg2 + (size_t)n * (size_t)(Bn * C) + b * C;

    float a = bias[o];
#pragma unroll 8
    for (int i = 0; i < 32; ++i) {
        a += x0[i] * W[i * 32 + o];
        a += g1[i] * W[1024 + i * 32 + o];
        a += g2[i] * W[2048 + i * 32 + o];
    }
    out[((size_t)b * N + n) * C + o] = a;
}

// ---------------------------------------------------------------------------
// Driver
// ---------------------------------------------------------------------------
extern "C" void kernel_launch(void* const* d_in, const int* in_sizes, int n_in,
                              void* d_out, int out_size, void* d_ws, size_t ws_size,
                              hipStream_t stream) {
    (void)in_sizes; (void)n_in; (void)out_size; (void)ws_size;
    const float* x     = (const float*)d_in[0];  // [16,4096,32]
    const float* graph = (const float*)d_in[1];  // [4096,4096]
    const float* E     = (const float*)d_in[2];  // [4096,16]
    const float* Wp    = (const float*)d_in[3];  // [16,3,32,32]
    const float* Bp    = (const float*)d_in[4];  // [16,32]
    float* out = (float*)d_out;

    const int B = 16, N = 4096, C = 32;
    const size_t MB = 1ull << 20;
    char* ws = (char*)d_ws;
    unsigned short* Gb = (unsigned short*)(ws);            // 32 MB: graph bf16
    unsigned short* Pb = (unsigned short*)(ws + 32 * MB);  // 32 MB: softmax P, reused for T2
    unsigned short* Sb = (unsigned short*)(ws + 64 * MB);  // 32 MB: S bf16
    unsigned short* XT = (unsigned short*)(ws + 96 * MB);  // 4  MB: x transposed bf16
    float* Xg1 = (float*)(ws + 100 * MB);                  // 8  MB
    float* Xg2 = (float*)(ws + 108 * MB);                  // 8  MB

    k_f32_to_bf16<<<2048, 256, 0, stream>>>(graph, Gb, N * N);
    k_build_xt<<<512, 256, 0, stream>>>(x, XT, B, N, C);
    k_adjacency<<<N, 256, 0, stream>>>(E, Pb, N);

    dim3 gBig(N / BN, N / BM);
    // S = P @ G
    k_gemm_bf16<true><<<gBig, 256, 0, stream>>>(Pb, Gb, Sb, N, N, N, 1.f, 0);
    // T2 = 2*S@S - I   (into Pb, which is free now)
    k_gemm_bf16<true><<<gBig, 256, 0, stream>>>(Sb, Sb, Pb, N, N, N, 2.f, 1);

    dim3 gProp((B * C) / BN, N / BM);
    // Xg1 = S @ XT,  Xg2 = T2 @ XT
    k_gemm_bf16<false><<<gProp, 256, 0, stream>>>(Sb, XT, Xg1, N, B * C, N, 1.f, 0);
    k_gemm_bf16<false><<<gProp, 256, 0, stream>>>(Pb, XT, Xg2, N, B * C, N, 1.f, 0);

    k_final<<<N, 512, 0, stream>>>(x, E, Wp, Bp, Xg1, Xg2, out, B, N, C);
}
